// TCN_4234837754224
// MI455X (gfx1250) — matrix operand order repaired and checked by reading, not compiled, not run on hardware
//
#include <hip/hip_runtime.h>
#include <cstdint>
#include <cstddef>

// ---------------- CDNA5 (gfx1250, wave32) WMMA types ----------------
typedef __attribute__((ext_vector_type(16))) __bf16 v16bf;
typedef __attribute__((ext_vector_type(8)))  float  v8f;

#define B_    16
#define T_    2048
#define IN_   34
#define C_    256
#define NCLS_ 5
#define BT_   (B_ * T_)          // 32768
#define BCT_  (B_ * C_ * T_)     // 8388608

// =====================================================================
// 1) Transpose input x[B,T,34] f32 -> act[B,256,T] bf16, rows 0..63 only (34..63 zero)
// =====================================================================
__global__ void k_transpose_in(const float* __restrict__ x, __bf16* __restrict__ act) {
    int idx = blockIdx.x * 256 + threadIdx.x;   // B*64*T = 2^21
    int t  = idx & (T_ - 1);
    int ci = (idx >> 11) & 63;
    int b  = idx >> 17;
    float v = (ci < IN_) ? x[((size_t)b * T_ + t) * IN_ + ci] : 0.0f;
    act[((size_t)(b * C_ + ci)) * T_ + t] = (__bf16)v;
}

// =====================================================================
// 2) Fold weight-norm (w = g*v/||v||) and pre-swizzle into WMMA A-matrix
//    lane layout (this file's contiguous-k order, matching B): lane=16*g+m ; K = e + 16g
//    tile = 1024B (32B/lane), tiles indexed [(mt*3+k)*cinTiles + ct]
// =====================================================================
__global__ void k_fold_w(const float* __restrict__ v, const float* __restrict__ g,
                         int cin, int cinTiles, __bf16* __restrict__ wsw) {
    int co = threadIdx.x;                      // 256 threads == Cout
    int kk = cin * 3;
    float nrm = 0.0f;
    for (int i = 0; i < kk; ++i) { float w = v[co * kk + i]; nrm += w * w; }
    float s = g[co] / sqrtf(nrm);
    int mt = co >> 4, m = co & 15;
    for (int k = 0; k < 3; ++k)
        for (int ct = 0; ct < cinTiles; ++ct)
            for (int kidx = 0; kidx < 32; ++kidx) {
                int ci = ct * 32 + kidx;
                float w = (ci < cin) ? v[(co * cin + ci) * 3 + k] * s : 0.0f;
                int gq   = kidx >> 4;
                int e    = kidx & 15;
                int lane = gq * 16 + m;
                wsw[(size_t)((mt * 3 + k) * cinTiles + ct) * 512 + lane * 16 + e] = (__bf16)w;
            }
}

// =====================================================================
// 3) Dilated causal conv as 3 shifted GEMMs: Y[b,co,t] = sum_k W_k @ X(t-k*d)
//    Block: 8 waves = 4 co-tiles x 2 t-halves; each wave holds FOUR
//    independent 16x16 f32 accumulators (4 t-tiles) so A-fragments are
//    reused 4x and the 12 WMMAs per K-chunk are hazard-independent.
//    LDS tile stored [t][ci] so B-fragments are 32B-contiguous per lane.
// =====================================================================
__global__ __launch_bounds__(256)
void k_conv_wmma(const __bf16* __restrict__ src, const __bf16* __restrict__ wsw,
                 float* __restrict__ dst, int cinTiles, int dil) {
    __shared__ __align__(32) __bf16 smem[192 * 32];     // [t_local][ci32] max width 128+2*32
    int tid = threadIdx.x, lane = tid & 31, w = tid >> 5;
    int bx = blockIdx.x;
    int b = bx >> 4, tb = bx & 15;
    int tstart = tb * 128;
    int Wl = 128 + 2 * dil;
    int gq = lane >> 4, n = lane & 15;
    int mtl = w >> 1;                 // local co-tile 0..3
    int th  = w & 1;                  // t-half 0..1 (64 t each)
    int mt  = blockIdx.y * 4 + mtl;   // global co-tile 0..15
    v8f acc[4];
#pragma unroll
    for (int tt = 0; tt < 4; ++tt) acc[tt] = (v8f){0.f,0.f,0.f,0.f,0.f,0.f,0.f,0.f};
    const v16bf* wv = (const v16bf*)wsw;

    for (int ct = 0; ct < cinTiles; ++ct) {
        __syncthreads();
        // stage X[b, ct*32 .. ct*32+31, tstart-2d .. tstart+127] -> smem[t][ci]
        for (int i = tid; i < 32 * Wl; i += 256) {
            int r = i / Wl, tl = i - r * Wl;
            int t = tstart - 2 * dil + tl;
            float val = 0.0f;
            if (t >= 0 && t < T_) val = (float)src[((size_t)(b * C_ + ct * 32 + r)) * T_ + t];
            smem[tl * 32 + r] = (__bf16)val;
        }
        if (ct + 1 < cinTiles)   // global_prefetch_b8 of next weight chunk
            __builtin_prefetch((const void*)(wv + (size_t)((mt * 3) * cinTiles + ct + 1) * 32 + lane), 0, 1);
        __syncthreads();
#pragma unroll
        for (int k = 0; k < 3; ++k) {
            v16bf a = wv[(size_t)((mt * 3 + k) * cinTiles + ct) * 32 + lane];
            int tl0 = 2 * dil + th * 64 + n - k * dil;         // in [0, Wl-48)
#pragma unroll
            for (int tt = 0; tt < 4; ++tt) {
                v16bf bm = *(const v16bf*)(&smem[(tl0 + tt * 16) * 32 + gq * 16]);
                acc[tt] = __builtin_amdgcn_wmma_f32_16x16x32_bf16(false, a, false, bm,
                                                                  (short)0, acc[tt], false, false);
            }
        }
    }
    // C/D layout: VGPR r, lanes 0-15 -> M=r, lanes 16-31 -> M=r+8 ; N = lane&15
#pragma unroll
    for (int tt = 0; tt < 4; ++tt) {
        int t = tstart + th * 64 + tt * 16 + n;
#pragma unroll
        for (int r = 0; r < 8; ++r) {
            int row = mt * 16 + r + 8 * gq;
            dst[((size_t)(b * C_ + row)) * T_ + t] = acc[tt][r];
        }
    }
}

// =====================================================================
// 4) BatchNorm batch statistics: per-channel sum / sumsq over (B,T)
// =====================================================================
__global__ void k_bn_stats(const float* __restrict__ y, float* __restrict__ stats) {
    int c = blockIdx.x, tid = threadIdx.x;
    __shared__ float s1[256], s2[256];
    float a = 0.f, q = 0.f;
    for (int i = tid; i < BT_; i += 256) {
        int b = i >> 11, t = i & (T_ - 1);
        float v = y[((size_t)(b * C_ + c)) * T_ + t];
        a += v; q += v * v;
    }
    s1[tid] = a; s2[tid] = q; __syncthreads();
    for (int s = 128; s > 0; s >>= 1) {
        if (tid < s) { s1[tid] += s1[tid + s]; s2[tid] += s2[tid + s]; }
        __syncthreads();
    }
    if (tid == 0) { stats[c] = s1[0]; stats[C_ + c] = s2[0]; }
}

// =====================================================================
// 5) BN apply + ReLU (-> bf16 next-conv input, or f32 in place)
// =====================================================================
template <bool TOBF>
__global__ void k_bn_apply(const float* __restrict__ y, const float* __restrict__ stats,
                           const float* __restrict__ gam, const float* __restrict__ bet,
                           float* __restrict__ outF, __bf16* __restrict__ outB) {
    size_t idx = (size_t)blockIdx.x * 256 + threadIdx.x;   // (b*C + c)*T + t
    int c = (int)((idx >> 11) & 255);
    float inv = 1.0f / (float)BT_;
    float m   = stats[c] * inv;
    float var = stats[C_ + c] * inv - m * m;
    float sc  = gam[c] * rsqrtf(var + 1e-5f);
    float v   = (y[idx] - m) * sc + bet[c];
    v = fmaxf(v, 0.0f);
    if (TOBF) outB[idx] = (__bf16)v; else outF[idx] = v;
}

// =====================================================================
// 6) SE: mean over T per (b,c)
// =====================================================================
__global__ void k_se_mean(const float* __restrict__ h, float* __restrict__ mean) {
    int c = blockIdx.x, b = blockIdx.y, tid = threadIdx.x;
    __shared__ float s1[256];
    const float* p = &h[((size_t)(b * C_ + c)) * T_];
    float a = 0.f;
    for (int t = tid; t < T_; t += 256) a += p[t];
    s1[tid] = a; __syncthreads();
    for (int s = 128; s > 0; s >>= 1) { if (tid < s) s1[tid] += s1[tid + s]; __syncthreads(); }
    if (tid == 0) mean[b * C_ + c] = s1[0] * (1.0f / T_);
}

// 7) SE FCs: relu(mean @ se1^T) -> sigmoid(. @ se2^T), one block.
__global__ void k_se_fc(const float* __restrict__ mean, const float* __restrict__ w1,
                        const float* __restrict__ w2, float* __restrict__ sev) {
    __shared__ float sm[B_ * C_];
    __shared__ float y1[B_ * 16];
    int tid = threadIdx.x;
    for (int i = tid; i < B_ * C_; i += 256) sm[i] = mean[i];
    __syncthreads();
    {
        int b = tid >> 4, r = tid & 15;
        float a = 0.f;
        for (int c = 0; c < C_; ++c) a += sm[b * C_ + c] * w1[r * C_ + c];
        y1[b * 16 + r] = fmaxf(a, 0.0f);
    }
    __syncthreads();
    for (int i = tid; i < B_ * C_; i += 256) {
        int b = i >> 8, c = i & 255;
        float a = 0.f;
#pragma unroll
        for (int r = 0; r < 16; ++r) a += y1[b * 16 + r] * w2[c * 16 + r];
        sev[i] = 1.0f / (1.0f + expf(-a));
    }
}

// =====================================================================
// 8) out = relu(h * se + residual); layer0 fuses the 1x1 downsample conv
// =====================================================================
template <bool DS>
__global__ void k_combine(const float* __restrict__ h, const float* __restrict__ sev,
                          const __bf16* __restrict__ resB, const float* __restrict__ dw,
                          const float* __restrict__ db, __bf16* __restrict__ outA) {
    size_t idx = (size_t)blockIdx.x * 256 + threadIdx.x;
    int t = (int)(idx & (T_ - 1));
    int c = (int)((idx >> 11) & 255);
    int b = (int)(idx >> 19);
    float res;
    if (DS) {
        float a = db[c];
        for (int ci = 0; ci < IN_; ++ci)
            a += dw[c * IN_ + ci] * (float)resB[((size_t)(b * C_ + ci)) * T_ + t];
        res = a;
    } else {
        res = (float)resB[idx];
    }
    float v = h[idx] * sev[b * C_ + c] + res;
    outA[idx] = (__bf16)fmaxf(v, 0.0f);
}

// =====================================================================
// 9) attention + logits + softmax + argmax
// =====================================================================
__global__ void k_logits(const __bf16* __restrict__ z, const float* __restrict__ Wa,
                         const float* __restrict__ ba, const float* __restrict__ Wl,
                         const float* __restrict__ bl, float* __restrict__ logits,
                         float* __restrict__ probs, int* __restrict__ preds) {
    __shared__ float wsh[6 * C_];
    int tid = threadIdx.x;
    for (int i = tid; i < 5 * C_; i += 256) wsh[i] = Wl[i];
    for (int i = tid; i < C_;      i += 256) wsh[5 * C_ + i] = Wa[i];
    __syncthreads();
    int gidx = blockIdx.x * 256 + tid;        // b*T + t
    int b = gidx >> 11, t = gidx & (T_ - 1);
    float acc[5] = {0, 0, 0, 0, 0}, aA = 0.f;
    for (int c = 0; c < C_; ++c) {
        float zv = (float)z[((size_t)(b * C_ + c)) * T_ + t];
        aA += zv * wsh[5 * C_ + c];
#pragma unroll
        for (int j = 0; j < 5; ++j) acc[j] += zv * wsh[j * C_ + c];
    }
    float attn = 1.0f / (1.0f + expf(-(aA + ba[0])));
    float lg[5], mx = -1e30f;
#pragma unroll
    for (int j = 0; j < 5; ++j) { lg[j] = attn * acc[j] + bl[j]; mx = fmaxf(mx, lg[j]); }
    float ex[5], sm = 0.f;
#pragma unroll
    for (int j = 0; j < 5; ++j) { ex[j] = expf(lg[j] - mx); sm += ex[j]; }
    float inv = 1.0f / sm;
    int am = 0; float bv = lg[0];
#pragma unroll
    for (int j = 0; j < 5; ++j) {
        logits[(size_t)gidx * 5 + j] = lg[j];
        probs[(size_t)gidx * 5 + j]  = ex[j] * inv;
        if (j > 0 && lg[j] > bv) { bv = lg[j]; am = j; }
    }
    preds[gidx] = am;
}

// =====================================================================
// 10) conservative no-activity filter (15-wide window, direct count)
// =====================================================================
__global__ void k_conservative(const int* __restrict__ preds, const int* __restrict__ flag,
                               int* __restrict__ proc) {
    int gidx = blockIdx.x * 256 + threadIdx.x;
    int b = gidx >> 11, t = gidx & (T_ - 1);
    int p = preds[b * T_ + t];
    int outv = p;
    if (flag[0] != 0) {
        int s = t - 7; if (s < 0) s = 0;
        int e = t + 8; if (e > T_) e = T_;
        int cnt[5] = {0, 0, 0, 0, 0};
        for (int u = s; u < e; ++u) cnt[preds[b * T_ + u]]++;
        int ws = e - s;
        int non3 = ws - cnt[3];
        float ratio = (float)non3 / (float)ws;
        int prevp = (t > 0)      ? preds[b * T_ + t - 1]  : preds[b * T_];
        int nextp = (t < T_ - 1) ? preds[b * T_ + t + 1]  : preds[b * T_ + T_ - 1];
        bool interior = (t > 0) && (t < T_ - 1);
        bool cond  = (p == 3) && (ratio > 0.4f) && interior;
        bool case1 = cond && (prevp == nextp) && (prevp != 3);
        int cw[5] = {cnt[0], cnt[1], cnt[2], -1, cnt[4]};
        int mode = 0, bvv = cw[0];
        for (int j = 1; j < 5; ++j) if (cw[j] > bvv) { bvv = cw[j]; mode = j; }
        bool case2 = cond && (!case1) && (non3 > 0);
        outv = case1 ? prevp : (case2 ? mode : p);
    }
    proc[b * T_ + t] = outv;
}

// =====================================================================
// 11) transition-rule scan (sequential over T; 16 independent batch rows)
// =====================================================================
__global__ void k_transition(const int* __restrict__ proc, const float* __restrict__ probs,
                             const int* __restrict__ flag, float* __restrict__ outp) {
    const float tm[25] = {1, 1, 1, 0.5f, 1,
                          1, 1, 0, 0.5f, 1,
                          0, 1, 1, 0.7f, 1,
                          1.5f, 1.5f, 0.7f, 1, 1,
                          1, 1, 1, 1, 1};
    int b = threadIdx.x;
    if (b >= B_) return;
    int ft = flag[0];
    int prev = proc[b * T_];
    outp[b * T_] = (float)prev;
    for (int t = 1; t < T_; ++t) {
        int curr = proc[b * T_ + t];
        int res = curr;
        if (ft != 0) {
            const float* row = &tm[prev * 5];
            if (row[curr] < 1.0f) {
                float vp[5]; float sm = 0.f;
#pragma unroll
                for (int j = 0; j < 5; ++j) {
                    float pv = probs[((size_t)(b * T_ + t)) * 5 + j];
                    vp[j] = (row[j] < 1.0f) ? 0.0f : pv;
                    sm += vp[j];
                }
                if (sm == 0.0f) res = prev;
                else {
                    int am = 0; float bv = vp[0];
                    for (int j = 1; j < 5; ++j) if (vp[j] > bv) { bv = vp[j]; am = j; }
                    res = am;
                }
            }
        }
        outp[b * T_ + t] = (float)res;
        prev = res;
    }
}

// =====================================================================
// Host side
// =====================================================================
extern "C" void kernel_launch(void* const* d_in, const int* in_sizes, int n_in,
                              void* d_out, int out_size, void* d_ws, size_t ws_size,
                              hipStream_t stream) {
    (void)in_sizes; (void)n_in; (void)ws_size;
    const float* X  = (const float*)d_in[0];
    const float* Wa = (const float*)d_in[1];
    const float* Wl = (const float*)d_in[2];
    const float* ba = (const float*)d_in[3];
    const float* bl = (const float*)d_in[4];
    struct LP { const float *b1,*b2,*bn1b,*bn1g,*bn2b,*bn2g,*db,*dw,*g1,*g2,*se1,*se2,*v1,*v2; };
    LP L[6];
    int idx = 5;
    for (int i = 0; i < 6; ++i) {
        L[i].b1   = (const float*)d_in[idx++];
        L[i].b2   = (const float*)d_in[idx++];
        L[i].bn1b = (const float*)d_in[idx++];
        L[i].bn1g = (const float*)d_in[idx++];
        L[i].bn2b = (const float*)d_in[idx++];
        L[i].bn2g = (const float*)d_in[idx++];
        if (i == 0) { L[i].db = (const float*)d_in[idx++]; L[i].dw = (const float*)d_in[idx++]; }
        else        { L[i].db = nullptr;                   L[i].dw = nullptr; }
        L[i].g1  = (const float*)d_in[idx++];
        L[i].g2  = (const float*)d_in[idx++];
        L[i].se1 = (const float*)d_in[idx++];
        L[i].se2 = (const float*)d_in[idx++];
        L[i].v1  = (const float*)d_in[idx++];
        L[i].v2  = (const float*)d_in[idx++];
    }
    const int* consFlag  = (const int*)d_in[idx++];
    const int* transFlag = (const int*)d_in[idx++];

    // workspace layout
    char* base = (char*)d_ws;
    __bf16* actA  = (__bf16*)(base + 0);                      // 16 MB
    __bf16* actB  = (__bf16*)(base + 16777216);               // 16 MB
    __bf16* actM  = (__bf16*)(base + 33554432);               // 16 MB
    float*  convY = (float*) (base + 50331648);               // 32 MB
    __bf16* wsw   = (__bf16*)(base + 83886080);               // 512 KB
    float*  stats = (float*) (base + 84410368);               // 2 KB
    float*  seMean= (float*) (base + 84414464);               // 16 KB
    float*  seVec = (float*) (base + 84431872);               // 16 KB
    float*  probs = (float*) (base + 84448256);               // 640 KB
    int*    preds = (int*)   (base + 85103616);               // 128 KB
    int*    proc  = (int*)   (base + 85234688);               // 128 KB

    dim3 blk(256);
    // input transpose + bf16 cast (+ zero-pad channels to 64)
    k_transpose_in<<<dim3(B_ * 64 * T_ / 256), blk, 0, stream>>>(X, actA);

    for (int i = 0; i < 6; ++i) {
        int dil = 1 << i;
        int cin      = (i == 0) ? IN_ : C_;
        int cinTiles = (i == 0) ? 2 : 8;
        __bf16* in  = (i & 1) ? actB : actA;
        __bf16* out = (i & 1) ? actA : actB;

        // conv1 + BN1 + ReLU
        k_fold_w<<<dim3(1), blk, 0, stream>>>(L[i].v1, L[i].g1, cin, cinTiles, wsw);
        k_conv_wmma<<<dim3(B_ * 16, 4), blk, 0, stream>>>(in, wsw, convY, cinTiles, dil);
        k_bn_stats<<<dim3(C_), blk, 0, stream>>>(convY, stats);
        k_bn_apply<true><<<dim3(BCT_ / 256), blk, 0, stream>>>(convY, stats, L[i].bn1g, L[i].bn1b,
                                                               nullptr, actM);
        // conv2 + BN2 + ReLU (f32 kept for SE)
        k_fold_w<<<dim3(1), blk, 0, stream>>>(L[i].v2, L[i].g2, C_, 8, wsw);
        k_conv_wmma<<<dim3(B_ * 16, 4), blk, 0, stream>>>(actM, wsw, convY, 8, dil);
        k_bn_stats<<<dim3(C_), blk, 0, stream>>>(convY, stats);
        k_bn_apply<false><<<dim3(BCT_ / 256), blk, 0, stream>>>(convY, stats, L[i].bn2g, L[i].bn2b,
                                                                convY, nullptr);
        // SE
        k_se_mean<<<dim3(C_, B_), blk, 0, stream>>>(convY, seMean);
        k_se_fc<<<dim3(1), blk, 0, stream>>>(seMean, L[i].se1, L[i].se2, seVec);
        // scale + residual + ReLU (-> next layer input, bf16)
        if (i == 0)
            k_combine<true><<<dim3(BCT_ / 256), blk, 0, stream>>>(convY, seVec, in, L[i].dw, L[i].db, out);
        else
            k_combine<false><<<dim3(BCT_ / 256), blk, 0, stream>>>(convY, seVec, in, nullptr, nullptr, out);
    }

    // final activations live in actA (layer 5 writes A)
    float* outLogits = (float*)d_out;
    k_logits<<<dim3(BT_ / 256), blk, 0, stream>>>(actA, Wa, ba, Wl, bl, outLogits, probs, preds);

    if (out_size >= BT_ * NCLS_ + BT_) {
        float* outProc = outLogits + (size_t)BT_ * NCLS_;
        k_conservative<<<dim3(BT_ / 256), blk, 0, stream>>>(preds, consFlag, proc);
        k_transition<<<dim3(1), dim3(32), 0, stream>>>(proc, probs, transFlag, outProc);
    }
}
